// VisionTransformerLGMLRP_80247168959030
// MI455X (gfx1250) — matrix operand order repaired and checked by reading, not compiled, not run on hardware
//
#include <hip/hip_runtime.h>
#include <math.h>

// ============================================================================
// ViT-Base forward on gfx1250 (MI455X).
// All GEMMs via v_wmma_f32_16x16x32_bf16 computing C^T = W^T * X^T so that
// BOTH operand fragments are contiguous ds_load_b128 reads and the epilogue
// is fully vectorized. Tile staging uses global_load_async_to_lds_b128
// (ASYNCcnt, in-order completion) with LDS double buffering so the async
// engine overlaps the WMMA stream.
// ============================================================================

typedef __attribute__((ext_vector_type(16))) __bf16 v16bf;
typedef __attribute__((ext_vector_type(8)))  float  v8f;

struct U4x2 { uint4 lo, hi; };   // 32 bytes == one v16bf fragment

__device__ __forceinline__ __bf16 f2bf(float f) {
  unsigned u = __builtin_bit_cast(unsigned, f);
  u += 0x7fffu + ((u >> 16) & 1u);               // round-to-nearest-even
  unsigned short h = (unsigned short)(u >> 16);
  return __builtin_bit_cast(__bf16, h);
}
__device__ __forceinline__ float bf2f(__bf16 h) {
  unsigned short us = __builtin_bit_cast(unsigned short, h);
  unsigned u = ((unsigned)us) << 16;
  return __builtin_bit_cast(float, u);
}
__device__ __forceinline__ unsigned pk2bf(float a, float b) {
  unsigned short ua = __builtin_bit_cast(unsigned short, f2bf(a));
  unsigned short ub = __builtin_bit_cast(unsigned short, f2bf(b));
  return (unsigned)ua | ((unsigned)ub << 16);
}
__device__ __forceinline__ float gelu_exact(float x) {
  return 0.5f * x * (1.0f + erff(x * 0.70710678118654752f));
}

// ---- CDNA5 async tile staging: global -> LDS without touching VGPRs -------
#define USE_ASYNC_LDS 1

__device__ __forceinline__ void tile_copy_b128(const __bf16* g, __bf16* l) {
#if USE_ASYNC_LDS
  // VDST = per-lane LDS byte address (low 32 bits of generic pointer),
  // VADDR = 64-bit global address, GV mode (saddr = off). Tracked by ASYNCcnt.
  asm volatile("global_load_async_to_lds_b128 %0, %1, off"
               :: "v"((unsigned)(uintptr_t)l),
                  "v"((unsigned long long)(uintptr_t)g)
               : "memory");
#else
  *reinterpret_cast<uint4*>(l) = *reinterpret_cast<const uint4*>(g);
#endif
}
// wait until at most N async ops remain in flight (in-order completion)
template <int N>
__device__ __forceinline__ void wait_async() {
#if USE_ASYNC_LDS
  asm volatile("s_wait_asynccnt %0" :: "i"(N) : "memory");
#endif
}

#define VIT_B     32
#define VIT_E     768
#define VIT_T     8192      // 32 * 256 tokens
#define VIT_NH    12
#define VIT_HD    64
#define VIT_BH    384       // B * H
#define VIT_MLP   3072
#define VIT_DEPTH 12

// ---------------------------------------------------------------------------
// elementwise converts
// ---------------------------------------------------------------------------
__global__ __launch_bounds__(256) void f32_to_bf16_kernel(
    const float* __restrict__ src, __bf16* __restrict__ dst, long long n) {
  long long i = (long long)blockIdx.x * 256 + threadIdx.x;
  if (i < n) dst[i] = f2bf(src[i]);
}

// batched weight transpose+convert: (L, K, N) f32 -> (L, N, K) bf16
__global__ __launch_bounds__(256) void wtrans_bf16_kernel(
    const float* __restrict__ src, __bf16* __restrict__ dst,
    int L, int K, int N) {
  long long i = (long long)blockIdx.x * 256 + threadIdx.x;
  long long per = (long long)K * N;
  if (i >= (long long)L * per) return;
  int l = (int)(i / per), r = (int)(i % per);
  int n = r / K, k = r % K;
  dst[(size_t)l * per + (size_t)n * K + k] =
      f2bf(src[(size_t)l * per + (size_t)k * N + n]);
}

// im2col: samples (B,3,256,256) -> patches (T, 768) bf16
__global__ __launch_bounds__(256) void im2col_kernel(
    const float* __restrict__ samples, __bf16* __restrict__ patches) {
  int i = blockIdx.x * 256 + threadIdx.x;
  if (i >= VIT_T * 768) return;
  int t  = i / 768, kf = i % 768;
  int b  = t >> 8,  n  = t & 255;
  int gy = n >> 4,  gx = n & 15;
  int c  = kf >> 8, r  = kf & 255;
  int py = r >> 4,  px = r & 15;
  size_t src = (((size_t)(b * 3 + c) * 256) + (size_t)(gy * 16 + py)) * 256
             + (size_t)(gx * 16 + px);
  patches[i] = f2bf(samples[src]);
}

// x = (conv_out + conv_b)*bn_g*rsqrt(1+eps) + bn_b + pos2d
__global__ __launch_bounds__(256) void embed_finalize_kernel(
    const float* __restrict__ g, const float* __restrict__ conv_b,
    const float* __restrict__ bn_g, const float* __restrict__ bn_b,
    float* __restrict__ x) {
  int i = blockIdx.x * 256 + threadIdx.x;
  if (i >= VIT_T * VIT_E) return;
  int t = i / VIT_E, e = i % VIT_E;
  int n = t & 255, gy = n >> 4, gx = n & 15;
  float scale = bn_g[e] * rsqrtf(1.0f + 1e-5f);
  int   j    = (e < 384) ? e : e - 384;
  float posv = (e < 384) ? (float)(gy + 1) : (float)(gx + 1);
  float freq = powf(10000.0f, (2.0f * (float)(j >> 1)) / 384.0f);
  float arg  = posv / freq;
  float pos  = (j & 1) ? cosf(arg) : sinf(arg);
  x[i] = (g[i] + conv_b[e]) * scale + bn_b[e] + pos;
}

// ---------------------------------------------------------------------------
// WMMA GEMM (operand-swapped): out[m][n] = sum_k X[m][k] * Wt[n][k]
//   X : M x K row-major bf16 (activations)     -> B operand (interleaved K)
//   Wt: N x K row-major bf16 (transposed wts)  -> A operand (interleaved K)
// D layout: lane = m-offset, VGPR r = n-offset -> 8 consecutive N per lane.
// Double-buffered LDS; async loads for tile k+1 overlap WMMA on tile k.
// ---------------------------------------------------------------------------
template <bool HAS_BIAS, bool HAS_RES, bool DO_GELU, bool OUT_BF16>
__global__ __launch_bounds__(256) void gemm_wmma_kernel(
    const __bf16* __restrict__ X, const __bf16* __restrict__ Wt,
    const float* __restrict__ bias, const float* __restrict__ res,
    float* __restrict__ Cf, __bf16* __restrict__ Cb, int M, int N, int K) {
  __shared__ __align__(16) __bf16 Xs[2][128][32];
  __shared__ __align__(16) __bf16 Ws[2][128][32];
  const int tid = threadIdx.x, lane = tid & 31, wid = tid >> 5;
  const int wmw = wid & 3;                 // m quadrant: 32 rows (2 tiles)
  const int wnw = wid >> 2;                // n half   : 64 cols (4 tiles)
  const int bm = blockIdx.y * 128, bn = blockIdx.x * 128;
  const int lr = lane & 15, kh = lane >> 4;

  auto stage = [&](int buf, int k0) {      // 4 async b128 per thread
#pragma unroll
    for (int it = 0; it < 2; ++it) {
      int v = tid + it * 256, r = v >> 2, c4 = v & 3;
      tile_copy_b128(X  + (size_t)(bm + r) * K + (k0 + c4 * 8),
                     &Xs[buf][r][c4 * 8]);
      tile_copy_b128(Wt + (size_t)(bn + r) * K + (k0 + c4 * 8),
                     &Ws[buf][r][c4 * 8]);
    }
  };

  v8f acc[4][2];
#pragma unroll
  for (int a = 0; a < 4; ++a)
#pragma unroll
    for (int b = 0; b < 2; ++b)
#pragma unroll
      for (int r = 0; r < 8; ++r) acc[a][b][r] = 0.0f;

  stage(0, 0);
  const int nk = K >> 5;
  for (int ki = 0; ki < nk; ++ki) {
    const int cur = ki & 1;
    if (ki + 1 < nk) {                     // prefetch next tile, then wait for
      stage(cur ^ 1, (ki + 1) * 32);       // current tile only (<=4 in flight)
      wait_async<4>();
    } else {
      wait_async<0>();
    }
    __syncthreads();

    v16bf wf[4];
#pragma unroll
    for (int tn = 0; tn < 4; ++tn) {       // A operand: interleaved K halves
      int n = wnw * 64 + tn * 16 + lr;
      U4x2 t2;
      t2.lo = *reinterpret_cast<const uint4*>(&Ws[cur][n][kh * 8]);
      t2.hi = *reinterpret_cast<const uint4*>(&Ws[cur][n][16 + kh * 8]);
      wf[tn] = __builtin_bit_cast(v16bf, t2);
    }
    v16bf xf[2];
#pragma unroll
    for (int tm = 0; tm < 2; ++tm) {       // B operand: interleaved K halves
      int m = wmw * 32 + tm * 16 + lr;
      U4x2 t2;
      t2.lo = *reinterpret_cast<const uint4*>(&Xs[cur][m][kh * 8]);
      t2.hi = *reinterpret_cast<const uint4*>(&Xs[cur][m][16 + kh * 8]);
      xf[tm] = __builtin_bit_cast(v16bf, t2);
    }
#pragma unroll
    for (int tn = 0; tn < 4; ++tn)
#pragma unroll
      for (int tm = 0; tm < 2; ++tm)
        acc[tn][tm] = __builtin_amdgcn_wmma_f32_16x16x32_bf16(
            false, wf[tn], false, xf[tm], (short)0, acc[tn][tm], false, false);
    __syncthreads();                       // buffer-reuse safety
  }

#pragma unroll
  for (int tn = 0; tn < 4; ++tn) {
#pragma unroll
    for (int tm = 0; tm < 2; ++tm) {
      const int m  = bm + wmw * 32 + tm * 16 + lr;
      const int nb = bn + wnw * 64 + tn * 16 + ((lane < 16) ? 0 : 8);
      float v[8];
#pragma unroll
      for (int r = 0; r < 8; ++r) v[r] = acc[tn][tm][r];
      if (HAS_BIAS) {
        float4 b0 = *reinterpret_cast<const float4*>(&bias[nb]);
        float4 b1 = *reinterpret_cast<const float4*>(&bias[nb + 4]);
        v[0] += b0.x; v[1] += b0.y; v[2] += b0.z; v[3] += b0.w;
        v[4] += b1.x; v[5] += b1.y; v[6] += b1.z; v[7] += b1.w;
      }
      if (HAS_RES) {
        const float* rp = &res[(size_t)m * N + nb];
        float4 r0 = *reinterpret_cast<const float4*>(rp);
        float4 r1 = *reinterpret_cast<const float4*>(rp + 4);
        v[0] += r0.x; v[1] += r0.y; v[2] += r0.z; v[3] += r0.w;
        v[4] += r1.x; v[5] += r1.y; v[6] += r1.z; v[7] += r1.w;
      }
      if (DO_GELU) {
#pragma unroll
        for (int r = 0; r < 8; ++r) v[r] = gelu_exact(v[r]);
      }
      if (OUT_BF16) {
        uint4 o;
        o.x = pk2bf(v[0], v[1]); o.y = pk2bf(v[2], v[3]);
        o.z = pk2bf(v[4], v[5]); o.w = pk2bf(v[6], v[7]);
        *reinterpret_cast<uint4*>(&Cb[(size_t)m * N + nb]) = o;
      } else {
        float* dst = &Cf[(size_t)m * N + nb];
        float4 o0 = {v[0], v[1], v[2], v[3]};
        float4 o1 = {v[4], v[5], v[6], v[7]};
        *reinterpret_cast<float4*>(dst) = o0;
        *reinterpret_cast<float4*>(dst + 4) = o1;
      }
    }
  }
}

// ---------------------------------------------------------------------------
// qkv (T,2304) f32 -> q,k: (BH,256,64) bf16 ; v: head-transposed (BH,64,256)
// ---------------------------------------------------------------------------
__global__ __launch_bounds__(256) void split_heads_kernel(
    const float* __restrict__ qkv, __bf16* __restrict__ qh,
    __bf16* __restrict__ kh, __bf16* __restrict__ vt) {
  long long i = (long long)blockIdx.x * 256 + threadIdx.x;
  if (i >= (long long)VIT_T * 2304) return;
  int t = (int)(i / 2304), r = (int)(i % 2304);
  int s = r / 768, rem = r % 768, h = rem >> 6, d = rem & 63;
  int b = t >> 8, n = t & 255;
  __bf16 val = f2bf(qkv[i]);
  size_t bhI = (size_t)(b * VIT_NH + h);
  if (s == 0)      qh[(bhI * 256 + n) * 64 + d] = val;
  else if (s == 1) kh[(bhI * 256 + n) * 64 + d] = val;
  else             vt[(bhI * 64 + d) * 256 + n] = val;   // transposed
}

// ---------------------------------------------------------------------------
// S[m][n] = scale * Q[m][:].K[n][:]  per (b,h); swapped: A=K rows, B=Q rows.
// ---------------------------------------------------------------------------
__global__ __launch_bounds__(256) void attn_scores_kernel(
    const __bf16* __restrict__ Q, const __bf16* __restrict__ Kh,
    float* __restrict__ S, float scale) {
  __shared__ __align__(16) __bf16 Qs[128][64];
  __shared__ __align__(16) __bf16 Ks[128][64];
  const int bh = blockIdx.z;
  const __bf16* q  = Q  + (size_t)bh * 256 * 64;
  const __bf16* kk = Kh + (size_t)bh * 256 * 64;
  float* s = S + (size_t)bh * 256 * 256;
  const int tid = threadIdx.x, lane = tid & 31, wid = tid >> 5;
  const int wmw = wid & 3, wnw = wid >> 2, lr = lane & 15, kh = lane >> 4;
  const int bm = blockIdx.y * 128, bn = blockIdx.x * 128;

#pragma unroll
  for (int it = 0; it < 4; ++it) {
    int v = tid + it * 256, r = v >> 3, c4 = v & 7;
    tile_copy_b128(q  + (size_t)(bm + r) * 64 + c4 * 8, &Qs[r][c4 * 8]);
    tile_copy_b128(kk + (size_t)(bn + r) * 64 + c4 * 8, &Ks[r][c4 * 8]);
  }
  wait_async<0>();
  __syncthreads();

  v8f acc[4][2];
#pragma unroll
  for (int a = 0; a < 4; ++a)
#pragma unroll
    for (int b = 0; b < 2; ++b)
#pragma unroll
      for (int r = 0; r < 8; ++r) acc[a][b][r] = 0.0f;

#pragma unroll
  for (int k0 = 0; k0 < 64; k0 += 32) {
    v16bf kf[4];
#pragma unroll
    for (int tn = 0; tn < 4; ++tn) {
      int n = wnw * 64 + tn * 16 + lr;
      U4x2 t2;
      t2.lo = *reinterpret_cast<const uint4*>(&Ks[n][k0 + kh * 8]);
      t2.hi = *reinterpret_cast<const uint4*>(&Ks[n][k0 + 16 + kh * 8]);
      kf[tn] = __builtin_bit_cast(v16bf, t2);
    }
    v16bf qf[2];
#pragma unroll
    for (int tm = 0; tm < 2; ++tm) {
      int m = wmw * 32 + tm * 16 + lr;
      U4x2 t2;
      t2.lo = *reinterpret_cast<const uint4*>(&Qs[m][k0 + kh * 8]);
      t2.hi = *reinterpret_cast<const uint4*>(&Qs[m][k0 + 16 + kh * 8]);
      qf[tm] = __builtin_bit_cast(v16bf, t2);
    }
#pragma unroll
    for (int tn = 0; tn < 4; ++tn)
#pragma unroll
      for (int tm = 0; tm < 2; ++tm)
        acc[tn][tm] = __builtin_amdgcn_wmma_f32_16x16x32_bf16(
            false, kf[tn], false, qf[tm], (short)0, acc[tn][tm], false, false);
  }

#pragma unroll
  for (int tn = 0; tn < 4; ++tn)
#pragma unroll
    for (int tm = 0; tm < 2; ++tm) {
      const int m  = bm + wmw * 32 + tm * 16 + lr;
      const int nb = bn + wnw * 64 + tn * 16 + ((lane < 16) ? 0 : 8);
      float* dst = &s[(size_t)m * 256 + nb];
      float4 o0 = {acc[tn][tm][0] * scale, acc[tn][tm][1] * scale,
                   acc[tn][tm][2] * scale, acc[tn][tm][3] * scale};
      float4 o1 = {acc[tn][tm][4] * scale, acc[tn][tm][5] * scale,
                   acc[tn][tm][6] * scale, acc[tn][tm][7] * scale};
      *reinterpret_cast<float4*>(dst) = o0;
      *reinterpret_cast<float4*>(dst + 4) = o1;
    }
}

// ---------------------------------------------------------------------------
// row softmax, row length 256 == blockDim; emit bf16 probabilities
// ---------------------------------------------------------------------------
__global__ __launch_bounds__(256) void softmax_row256_kernel(
    const float* __restrict__ S, __bf16* __restrict__ P) {
  int row = blockIdx.x, tid = threadIdx.x;
  __shared__ float red[256];
  float v = S[(size_t)row * 256 + tid];
  red[tid] = v; __syncthreads();
  for (int off = 128; off > 0; off >>= 1) {
    if (tid < off) red[tid] = fmaxf(red[tid], red[tid + off]);
    __syncthreads();
  }
  float m = red[0];
  float e = expf(v - m);
  __syncthreads();
  red[tid] = e; __syncthreads();
  for (int off = 128; off > 0; off >>= 1) {
    if (tid < off) red[tid] += red[tid + off];
    __syncthreads();
  }
  P[(size_t)row * 256 + tid] = f2bf(e / red[0]);
}

// ---------------------------------------------------------------------------
// Out[m][h*64+d] = sum_j P[m][j] V[j][d]; swapped: A = V^T rows (d), B = P rows.
// Vt: (BH, 64, 256). Double-buffered async staging over the 8 k-steps.
// ---------------------------------------------------------------------------
__global__ __launch_bounds__(256) void attn_av_kernel(
    const __bf16* __restrict__ P, const __bf16* __restrict__ Vt,
    __bf16* __restrict__ Out) {
  __shared__ __align__(16) __bf16 Ps[2][128][32];
  __shared__ __align__(16) __bf16 Vs[2][64][32];
  const int bh = blockIdx.z, b = bh / VIT_NH, h = bh % VIT_NH;
  const __bf16* p  = P  + (size_t)bh * 256 * 256;
  const __bf16* vt = Vt + (size_t)bh * 64 * 256;
  const int tid = threadIdx.x, lane = tid & 31, wid = tid >> 5;
  const int wmw = wid & 3;                 // m quadrant (32 tokens, 2 tiles)
  const int wdw = wid >> 2;                // d half (32 dims, 2 tiles)
  const int lr = lane & 15, kh = lane >> 4;
  const int bm = blockIdx.y * 128;

  auto stage = [&](int buf, int k0) {      // 3 async b128 per thread
#pragma unroll
    for (int it = 0; it < 2; ++it) {
      int v = tid + it * 256, r = v >> 2, c4 = v & 3;
      tile_copy_b128(p + (size_t)(bm + r) * 256 + (k0 + c4 * 8),
                     &Ps[buf][r][c4 * 8]);
    }
    int r = tid >> 2, c4 = tid & 3;
    tile_copy_b128(vt + (size_t)r * 256 + (k0 + c4 * 8), &Vs[buf][r][c4 * 8]);
  };

  v8f acc[2][2];
#pragma unroll
  for (int a = 0; a < 2; ++a)
#pragma unroll
    for (int c = 0; c < 2; ++c)
#pragma unroll
      for (int r = 0; r < 8; ++r) acc[a][c][r] = 0.0f;

  stage(0, 0);
  for (int ki = 0; ki < 8; ++ki) {         // K = 256, 32 per step
    const int cur = ki & 1;
    if (ki + 1 < 8) {
      stage(cur ^ 1, (ki + 1) * 32);
      wait_async<3>();
    } else {
      wait_async<0>();
    }
    __syncthreads();

    v16bf vf[2];
#pragma unroll
    for (int td = 0; td < 2; ++td) {       // A operand: V^T rows (d)
      int d = wdw * 32 + td * 16 + lr;
      U4x2 t2;
      t2.lo = *reinterpret_cast<const uint4*>(&Vs[cur][d][kh * 8]);
      t2.hi = *reinterpret_cast<const uint4*>(&Vs[cur][d][16 + kh * 8]);
      vf[td] = __builtin_bit_cast(v16bf, t2);
    }
    v16bf pf[2];
#pragma unroll
    for (int tm = 0; tm < 2; ++tm) {       // B operand: P rows (m)
      int m = wmw * 32 + tm * 16 + lr;
      U4x2 t2;
      t2.lo = *reinterpret_cast<const uint4*>(&Ps[cur][m][kh * 8]);
      t2.hi = *reinterpret_cast<const uint4*>(&Ps[cur][m][16 + kh * 8]);
      pf[tm] = __builtin_bit_cast(v16bf, t2);
    }
#pragma unroll
    for (int td = 0; td < 2; ++td)
#pragma unroll
      for (int tm = 0; tm < 2; ++tm)
        acc[td][tm] = __builtin_amdgcn_wmma_f32_16x16x32_bf16(
            false, vf[td], false, pf[tm], (short)0, acc[td][tm], false, false);
    __syncthreads();                       // buffer-reuse safety
  }

#pragma unroll
  for (int td = 0; td < 2; ++td)
#pragma unroll
    for (int tm = 0; tm < 2; ++tm) {
      const int m  = bm + wmw * 32 + tm * 16 + lr;        // token in image
      const int db = wdw * 32 + td * 16 + ((lane < 16) ? 0 : 8);
      uint4 o;
      o.x = pk2bf(acc[td][tm][0], acc[td][tm][1]);
      o.y = pk2bf(acc[td][tm][2], acc[td][tm][3]);
      o.z = pk2bf(acc[td][tm][4], acc[td][tm][5]);
      o.w = pk2bf(acc[td][tm][6], acc[td][tm][7]);
      *reinterpret_cast<uint4*>(
          &Out[(size_t)(b * 256 + m) * VIT_E + h * 64 + db]) = o;
    }
}

// ---------------------------------------------------------------------------
// LayerNorm over last dim (768), emits bf16 for the next GEMM
// ---------------------------------------------------------------------------
__global__ __launch_bounds__(256) void layernorm_bf16_kernel(
    const float* __restrict__ X, const float* __restrict__ g,
    const float* __restrict__ b, __bf16* __restrict__ Y) {
  int t = blockIdx.x, tid = threadIdx.x;
  const float* x = X + (size_t)t * VIT_E;
  __shared__ float r1[256], r2[256];
  float s = 0.0f, s2 = 0.0f;
  for (int c = tid; c < VIT_E; c += 256) { float v = x[c]; s += v; s2 += v * v; }
  r1[tid] = s; r2[tid] = s2; __syncthreads();
  for (int off = 128; off > 0; off >>= 1) {
    if (tid < off) { r1[tid] += r1[tid + off]; r2[tid] += r2[tid + off]; }
    __syncthreads();
  }
  float mean = r1[0] * (1.0f / VIT_E);
  float var  = r2[0] * (1.0f / VIT_E) - mean * mean;
  float rstd = rsqrtf(var + 1e-5f);
  for (int c = tid; c < VIT_E; c += 256)
    Y[(size_t)t * VIT_E + c] = f2bf((x[c] - mean) * rstd * g[c] + b[c]);
}

// ---------------------------------------------------------------------------
// pooled head: out[b] = mean_n( dot(xn[b,n,:], head_w) ) + head_b
// ---------------------------------------------------------------------------
__global__ __launch_bounds__(256) void pooled_head_kernel(
    const __bf16* __restrict__ Xn, const float* __restrict__ head_w,
    const float* __restrict__ head_b, float* __restrict__ out) {
  int b = blockIdx.x, tid = threadIdx.x;             // tid == token in image
  const __bf16* x = Xn + ((size_t)b * 256 + tid) * VIT_E;
  float dot = 0.0f;
  for (int c = 0; c < VIT_E; ++c) dot += bf2f(x[c]) * head_w[c];
  __shared__ float red[256];
  red[tid] = dot; __syncthreads();
  for (int off = 128; off > 0; off >>= 1) {
    if (tid < off) red[tid] += red[tid + off];
    __syncthreads();
  }
  if (tid == 0) out[b] = red[0] * (1.0f / 256.0f) + head_b[0];
}

// ===========================================================================
extern "C" void kernel_launch(void* const* d_in, const int* in_sizes, int n_in,
                              void* d_out, int out_size, void* d_ws, size_t ws_size,
                              hipStream_t stream) {
  (void)in_sizes; (void)n_in; (void)out_size; (void)ws_size;
  const float* samples = (const float*)d_in[0];
  const float* conv_w  = (const float*)d_in[1];
  const float* conv_b  = (const float*)d_in[2];
  const float* bn_g    = (const float*)d_in[3];
  const float* bn_b    = (const float*)d_in[4];
  const float* ln1_g   = (const float*)d_in[5];
  const float* ln1_b   = (const float*)d_in[6];
  const float* qkv_w   = (const float*)d_in[7];
  const float* proj_w  = (const float*)d_in[8];
  const float* proj_b  = (const float*)d_in[9];
  const float* ln2_g   = (const float*)d_in[10];
  const float* ln2_b   = (const float*)d_in[11];
  const float* fc1_w   = (const float*)d_in[12];
  const float* fc1_b   = (const float*)d_in[13];
  const float* fc2_w   = (const float*)d_in[14];
  const float* fc2_b   = (const float*)d_in[15];
  const float* lnf_g   = (const float*)d_in[16];
  const float* lnf_b   = (const float*)d_in[17];
  const float* head_w  = (const float*)d_in[18];
  const float* head_b  = (const float*)d_in[19];
  float* out = (float*)d_out;

  // ---- workspace layout (256B aligned bump allocator) ----
  char* ws = (char*)d_ws;
  size_t off = 0;
  auto alloc = [&](size_t bytes) -> void* {
    void* p = ws + off;
    off = (off + bytes + 255) & ~(size_t)255;
    return p;
  };
  const long long nWqkv = 12LL * 768 * 2304;
  const long long nWprj = 12LL * 768 * 768;
  const long long nWfc1 = 12LL * 768 * 3072;
  const long long nWfc2 = 12LL * 3072 * 768;
  __bf16* wqkv    = (__bf16*)alloc(nWqkv * 2);   // (12, 2304, 768) = N x K
  __bf16* wproj   = (__bf16*)alloc(nWprj * 2);   // (12, 768, 768)
  __bf16* wfc1    = (__bf16*)alloc(nWfc1 * 2);   // (12, 3072, 768)
  __bf16* wfc2    = (__bf16*)alloc(nWfc2 * 2);   // (12, 768, 3072)
  __bf16* wconv   = (__bf16*)alloc((size_t)768 * 768 * 2);  // native N x K
  __bf16* patches = (__bf16*)alloc((size_t)VIT_T * 768 * 2);
  float*  x       = (float*) alloc((size_t)VIT_T * VIT_E * 4);
  __bf16* xn      = (__bf16*)alloc((size_t)VIT_T * VIT_E * 2);
  float*  buf1    = (float*) alloc((size_t)VIT_T * VIT_MLP * 4);  // qkv / scores
  __bf16* buf2    = (__bf16*)alloc((size_t)VIT_T * VIT_MLP * 2);  // probs / gelu
  __bf16* qh      = (__bf16*)alloc((size_t)VIT_T * VIT_E * 2);
  __bf16* kh      = (__bf16*)alloc((size_t)VIT_T * VIT_E * 2);
  __bf16* vt      = (__bf16*)alloc((size_t)VIT_T * VIT_E * 2);    // (BH,64,256)
  __bf16* attn_o  = (__bf16*)alloc((size_t)VIT_T * VIT_E * 2);

  auto blocks = [](long long n) { return (unsigned)((n + 255) / 256); };
  enum GMode { GM_PLAIN, GM_BIAS_RES, GM_BIAS_GELU_BF16 };
  auto gemm = [&](GMode mode, const __bf16* X, const __bf16* Wt,
                  const float* bias, const float* res,
                  float* Cf, __bf16* Cb, int M, int N, int K) {
    dim3 grid(N / 128, M / 128);
    switch (mode) {
      case GM_PLAIN:
        gemm_wmma_kernel<false, false, false, false>
            <<<grid, 256, 0, stream>>>(X, Wt, bias, res, Cf, Cb, M, N, K);
        break;
      case GM_BIAS_RES:
        gemm_wmma_kernel<true, true, false, false>
            <<<grid, 256, 0, stream>>>(X, Wt, bias, res, Cf, Cb, M, N, K);
        break;
      case GM_BIAS_GELU_BF16:
        gemm_wmma_kernel<true, false, true, true>
            <<<grid, 256, 0, stream>>>(X, Wt, bias, res, Cf, Cb, M, N, K);
        break;
    }
  };

  // ---- one-shot converts (weights -> N x K bf16) ----
  wtrans_bf16_kernel<<<blocks(nWqkv), 256, 0, stream>>>(qkv_w, wqkv, 12, 768, 2304);
  wtrans_bf16_kernel<<<blocks(nWprj), 256, 0, stream>>>(proj_w, wproj, 12, 768, 768);
  wtrans_bf16_kernel<<<blocks(nWfc1), 256, 0, stream>>>(fc1_w, wfc1, 12, 768, 3072);
  wtrans_bf16_kernel<<<blocks(nWfc2), 256, 0, stream>>>(fc2_w, wfc2, 12, 3072, 768);
  f32_to_bf16_kernel<<<blocks(768LL * 768), 256, 0, stream>>>(conv_w, wconv,
                                                              768LL * 768);
  im2col_kernel<<<blocks((long long)VIT_T * 768), 256, 0, stream>>>(samples, patches);

  // ---- patch embedding: patches(T,768) x wconv(768,768 NxK) -> x ----
  gemm(GM_PLAIN, patches, wconv, nullptr, nullptr, buf1, nullptr, VIT_T, VIT_E, 768);
  embed_finalize_kernel<<<blocks((long long)VIT_T * VIT_E), 256, 0, stream>>>(
      buf1, conv_b, bn_g, bn_b, x);

  // ---- transformer layers ----
  const float att_scale = 0.125f;  // 1/sqrt(64)
  for (int i = 0; i < VIT_DEPTH; ++i) {
    layernorm_bf16_kernel<<<VIT_T, 256, 0, stream>>>(x, ln1_g + i * VIT_E,
                                                     ln1_b + i * VIT_E, xn);
    gemm(GM_PLAIN, xn, wqkv + (size_t)i * 768 * 2304, nullptr, nullptr,
         buf1, nullptr, VIT_T, 2304, 768);
    split_heads_kernel<<<blocks((long long)VIT_T * 2304), 256, 0, stream>>>(
        buf1, qh, kh, vt);
    attn_scores_kernel<<<dim3(2, 2, VIT_BH), 256, 0, stream>>>(qh, kh, buf1,
                                                               att_scale);
    softmax_row256_kernel<<<VIT_BH * 256, 256, 0, stream>>>(buf1, buf2);
    attn_av_kernel<<<dim3(1, 2, VIT_BH), 256, 0, stream>>>(buf2, vt, attn_o);
    gemm(GM_BIAS_RES, attn_o, wproj + (size_t)i * 768 * 768,
         proj_b + i * VIT_E, x, x, nullptr, VIT_T, VIT_E, 768);
    layernorm_bf16_kernel<<<VIT_T, 256, 0, stream>>>(x, ln2_g + i * VIT_E,
                                                     ln2_b + i * VIT_E, xn);
    gemm(GM_BIAS_GELU_BF16, xn, wfc1 + (size_t)i * 768 * 3072,
         fc1_b + i * VIT_MLP, nullptr, nullptr, buf2, VIT_T, VIT_MLP, 768);
    gemm(GM_BIAS_RES, buf2, wfc2 + (size_t)i * 3072 * 768,
         fc2_b + i * VIT_E, x, x, nullptr, VIT_T, VIT_E, 3072);
  }

  // ---- final LN + pooled head ----
  layernorm_bf16_kernel<<<VIT_T, 256, 0, stream>>>(x, lnf_g, lnf_b, xn);
  pooled_head_kernel<<<VIT_B, 256, 0, stream>>>(xn, head_w, head_b, out);
}
